// TransformerBlock_78838419685611
// MI455X (gfx1250) — compile-verified, matrix operand order checked by reading, not run on hardware
//
#include <hip/hip_runtime.h>
#include <hip/hip_bf16.h>

typedef __bf16 bf16;
typedef bf16  v16bf __attribute__((ext_vector_type(16)));
typedef float v8f   __attribute__((ext_vector_type(8)));

#define D_MODEL   1024
#define NUM_HEADS 16
#define DK        64
#define D_FF      4096
#define BATCH     2
#define SEQ       2048
#define EPS       1e-5f

union Pack8 { uint4 q; bf16 h[8]; };
union Pack4 { uint2 d; bf16 h[4]; };
union AccFrag { v8f v; float f[8]; };
union ABFrag  { v16bf v; unsigned u[8]; };

// ---------------------------------------------------------------- f32 -> bf16
__global__ __launch_bounds__(256) void cvt_f32_bf16(const float* __restrict__ in,
                                                    bf16* __restrict__ out, int n) {
  int i = (blockIdx.x * 256 + threadIdx.x) * 4;
  if (i + 3 < n) {
    float4 f = *(const float4*)(in + i);
    Pack4 p;
    p.h[0] = (bf16)f.x; p.h[1] = (bf16)f.y; p.h[2] = (bf16)f.z; p.h[3] = (bf16)f.w;
    *(uint2*)(out + i) = p.d;
  }
}

// ------------------------------------------- transpose+convert W[K][N] -> WT[N][K]
__global__ __launch_bounds__(256)
void transpose_cvt_w(const float* __restrict__ W, bf16* __restrict__ WT, int Kd, int Nd) {
  __shared__ __align__(16) bf16 T[32][36];
  const int t  = threadIdx.x;
  const int n0 = blockIdx.x * 32;
  const int k0 = blockIdx.y * 32;
  {
    const int kk = t >> 3, nn = (t & 7) * 4;
    float4 f = *(const float4*)(W + (long long)(k0 + kk) * Nd + n0 + nn);
    Pack4 p; p.h[0] = (bf16)f.x; p.h[1] = (bf16)f.y; p.h[2] = (bf16)f.z; p.h[3] = (bf16)f.w;
    *(uint2*)&T[kk][nn] = p.d;
  }
  __syncthreads();
  {
    const int nn = t >> 3, ks = (t & 7) * 4;
    Pack4 p;
    p.h[0] = T[ks][nn]; p.h[1] = T[ks + 1][nn]; p.h[2] = T[ks + 2][nn]; p.h[3] = T[ks + 3][nn];
    *(uint2*)(WT + (long long)(n0 + nn) * Kd + k0 + ks) = p.d;
  }
}

// ------------------------------------------- v[B,S,D] -> vT[B,H,64,S] (bf16)
__global__ __launch_bounds__(256)
void transpose_v(const bf16* __restrict__ v, bf16* __restrict__ vT) {
  __shared__ __align__(16) bf16 T[32][36];
  const int t  = threadIdx.x;
  const int s0 = blockIdx.x * 32;
  const int d0 = blockIdx.y * 32;
  const int bh = blockIdx.z;            // b*16+h
  const int b  = bh >> 4, h = bh & 15;
  {
    const int ss = t >> 3, dd = (t & 7) * 4;
    const bf16* src = v + (long long)(b * SEQ + s0 + ss) * D_MODEL + h * 64 + d0 + dd;
    *(uint2*)&T[ss][dd] = *(const uint2*)src;
  }
  __syncthreads();
  {
    const int dd = t >> 3, ss = (t & 7) * 4;
    Pack4 p;
    p.h[0] = T[ss][dd]; p.h[1] = T[ss + 1][dd]; p.h[2] = T[ss + 2][dd]; p.h[3] = T[ss + 3][dd];
    bf16* dst = vT + (long long)(bh * 64 + d0 + dd) * SEQ + s0 + ss;
    *(uint2*)dst = p.d;
  }
}

// ---------------------------------------------------------------- WMMA GEMM
// C[M,N] = scale * (A[M,K] @ B^T) + bias;  B stored [N][K].  AF32: A is f32
// (softmaxed attn), converted to bf16 while staging.  Double-buffered LDS
// pipeline with named staging registers (no lambdas/arrays -> no scratch).
template<int BN, bool AF32, bool RELU>
__global__ __launch_bounds__(256)
void gemm_wmma(const void* __restrict__ Av, long long sAo, long long sAi, int lda,
               const bf16* __restrict__ B,  long long sBo, long long sBi, int ldb,
               const float* __restrict__ bias, float scale,
               float* __restrict__ Cf,  long long sCo,  long long sCi,  int ldc,
               bf16*  __restrict__ C16, long long sC16o, long long sC16i, int ldc16,
               int K)
{
  const int BM = 128, BK = 32;
  const int CF = BN / 32;   // wmma col-tiles per wave (wave tile 32 x BN/2)
  const int NB = BN / 64;   // B staging chunks per thread
  __shared__ __align__(16) bf16 As[2][BM][BK + 8];
  __shared__ __align__(16) bf16 Bs[2][BN][BK + 8];

  const int tid = threadIdx.x;
  const int z   = blockIdx.z;
  const long long aBase = (long long)(z >> 4) * sAo + (long long)(z & 15) * sAi;
  const long long bBase = (long long)(z >> 4) * sBo + (long long)(z & 15) * sBi;

  const int rowBlock = blockIdx.y * BM;
  const int colBlock = blockIdx.x * BN;

  const int st_r  = tid >> 2;        // 0..63
  const int st_ks = (tid & 3) * 8;   // 0/8/16/24

  // named per-thread global pointers (no arrays -> clean SROA)
  const float* aF0 = nullptr; const float* aF1 = nullptr;
  const bf16*  aB0 = nullptr; const bf16*  aB1 = nullptr;
  if (AF32) {
    const float* base = (const float*)Av + aBase;
    aF0 = base + (long long)(rowBlock + st_r) * lda + st_ks;
    aF1 = base + (long long)(rowBlock + st_r + 64) * lda + st_ks;
  } else {
    const bf16* base = (const bf16*)Av + aBase;
    aB0 = base + (long long)(rowBlock + st_r) * lda + st_ks;
    aB1 = base + (long long)(rowBlock + st_r + 64) * lda + st_ks;
  }
  const bf16* bP0 = B + bBase + (long long)(colBlock + st_r) * ldb + st_ks;
  const bf16* bP1 = (NB == 2) ? (B + bBase + (long long)(colBlock + st_r + 64) * ldb + st_ks)
                              : nullptr;

  uint4 ra0, ra1, rb0, rb1 = uint4{0, 0, 0, 0};

  // ---- tile 0 into buffer 0
  if (AF32) {
    float4 f0 = *(const float4*)aF0, f1 = *(const float4*)(aF0 + 4);
    float4 g0 = *(const float4*)aF1, g1 = *(const float4*)(aF1 + 4);
    Pack8 p, q;
    p.h[0] = (bf16)f0.x; p.h[1] = (bf16)f0.y; p.h[2] = (bf16)f0.z; p.h[3] = (bf16)f0.w;
    p.h[4] = (bf16)f1.x; p.h[5] = (bf16)f1.y; p.h[6] = (bf16)f1.z; p.h[7] = (bf16)f1.w;
    q.h[0] = (bf16)g0.x; q.h[1] = (bf16)g0.y; q.h[2] = (bf16)g0.z; q.h[3] = (bf16)g0.w;
    q.h[4] = (bf16)g1.x; q.h[5] = (bf16)g1.y; q.h[6] = (bf16)g1.z; q.h[7] = (bf16)g1.w;
    ra0 = p.q; ra1 = q.q; aF0 += BK; aF1 += BK;
  } else {
    ra0 = *(const uint4*)aB0; ra1 = *(const uint4*)aB1; aB0 += BK; aB1 += BK;
  }
  rb0 = *(const uint4*)bP0; bP0 += BK;
  if (NB == 2) { rb1 = *(const uint4*)bP1; bP1 += BK; }

  *(uint4*)&As[0][st_r][st_ks]      = ra0;
  *(uint4*)&As[0][st_r + 64][st_ks] = ra1;
  *(uint4*)&Bs[0][st_r][st_ks]      = rb0;
  if (NB == 2) *(uint4*)&Bs[0][st_r + 64][st_ks] = rb1;
  __syncthreads();

  const int wave = tid >> 5, lane = tid & 31;
  const int wrow = wave >> 1;       // 0..3  (32 rows each)
  const int wcol = wave & 1;        // 0..1  (BN/2 cols each)
  const int hl   = lane >> 4;
  const int idx  = lane & 15;

  AccFrag acc[2][CF];
  #pragma unroll
  for (int r = 0; r < 2; ++r)
    #pragma unroll
    for (int c = 0; c < CF; ++c)
      #pragma unroll
      for (int i = 0; i < 8; ++i) acc[r][c].f[i] = 0.f;

  int cur = 0;
  for (int k0 = 0;;) {
    const bool more = (k0 + BK) < K;
    if (more) {                               // issue next tile's global loads early
      if (AF32) {
        float4 f0 = *(const float4*)aF0, f1 = *(const float4*)(aF0 + 4);
        float4 g0 = *(const float4*)aF1, g1 = *(const float4*)(aF1 + 4);
        Pack8 p, q;
        p.h[0] = (bf16)f0.x; p.h[1] = (bf16)f0.y; p.h[2] = (bf16)f0.z; p.h[3] = (bf16)f0.w;
        p.h[4] = (bf16)f1.x; p.h[5] = (bf16)f1.y; p.h[6] = (bf16)f1.z; p.h[7] = (bf16)f1.w;
        q.h[0] = (bf16)g0.x; q.h[1] = (bf16)g0.y; q.h[2] = (bf16)g0.z; q.h[3] = (bf16)g0.w;
        q.h[4] = (bf16)g1.x; q.h[5] = (bf16)g1.y; q.h[6] = (bf16)g1.z; q.h[7] = (bf16)g1.w;
        ra0 = p.q; ra1 = q.q; aF0 += BK; aF1 += BK;
      } else {
        ra0 = *(const uint4*)aB0; ra1 = *(const uint4*)aB1; aB0 += BK; aB1 += BK;
      }
      rb0 = *(const uint4*)bP0; bP0 += BK;
      if (NB == 2) { rb1 = *(const uint4*)bP1; bP1 += BK; }
    }

    ABFrag fa[2], fb[CF];
    #pragma unroll
    for (int r = 0; r < 2; ++r) {
      const int row = wrow * 32 + r * 16 + idx;
      #pragma unroll
      for (int v = 0; v < 8; ++v) {
        const int k2 = ((v < 4) ? 0 : 16) + hl * 8 + (v & 3) * 2;
        fa[r].u[v] = *(const unsigned*)&As[cur][row][k2];
      }
    }
    #pragma unroll
    for (int c = 0; c < CF; ++c) {
      const int col = wcol * (BN / 2) + c * 16 + idx;
      #pragma unroll
      for (int v = 0; v < 8; ++v) {
        const int k2 = ((v < 4) ? 0 : 16) + hl * 8 + (v & 3) * 2;
        fb[c].u[v] = *(const unsigned*)&Bs[cur][col][k2];
      }
    }
    #pragma unroll
    for (int r = 0; r < 2; ++r)
      #pragma unroll
      for (int c = 0; c < CF; ++c)
        acc[r][c].v = __builtin_amdgcn_wmma_f32_16x16x32_bf16(
            false, fa[r].v, false, fb[c].v, (short)0, acc[r][c].v, false, false);

    if (!more) break;
    const int nxt = cur ^ 1;                  // other buffer: no race with reads of cur
    *(uint4*)&As[nxt][st_r][st_ks]      = ra0;
    *(uint4*)&As[nxt][st_r + 64][st_ks] = ra1;
    *(uint4*)&Bs[nxt][st_r][st_ks]      = rb0;
    if (NB == 2) *(uint4*)&Bs[nxt][st_r + 64][st_ks] = rb1;
    __syncthreads();
    cur = nxt; k0 += BK;
  }

  const long long cBase   = (long long)(z >> 4) * sCo   + (long long)(z & 15) * sCi;
  const long long c16Base = (long long)(z >> 4) * sC16o + (long long)(z & 15) * sC16i;
  #pragma unroll
  for (int r = 0; r < 2; ++r) {
    #pragma unroll
    for (int c = 0; c < CF; ++c) {
      const int n = colBlock + wcol * (BN / 2) + c * 16 + idx;
      const float bval = bias ? bias[n] : 0.f;
      #pragma unroll
      for (int i = 0; i < 8; ++i) {
        const int m = rowBlock + wrow * 32 + r * 16 + hl * 8 + i;
        float val = acc[r][c].f[i] * scale + bval;
        if (RELU) val = fmaxf(val, 0.f);
        if (Cf)  Cf[cBase + (long long)m * ldc + n] = val;
        if (C16) C16[c16Base + (long long)m * ldc16 + n] = (bf16)val;
      }
    }
  }
}

// ---------------------------------------------------------------- softmax
__global__ __launch_bounds__(256)
void softmax_rows(float* __restrict__ attn, const int* __restrict__ mask) {
  __shared__ float red[256];
  const int row = blockIdx.x;          // 0 .. B*H*S-1
  const int b   = row >> 15;           // / (H*S)
  float* p = attn + (long long)row * SEQ;
  const int* mrow = mask + b * SEQ;
  const int tid = threadIdx.x;
  float v[8];
  float mx = -3.4e38f;
  #pragma unroll
  for (int j = 0; j < 8; ++j) {
    const int i = tid + j * 256;
    float x = p[i];
    if (mrow[i] == 0) x = -1e9f;
    v[j] = x;
    mx = fmaxf(mx, x);
  }
  red[tid] = mx; __syncthreads();
  for (int s = 128; s > 0; s >>= 1) { if (tid < s) red[tid] = fmaxf(red[tid], red[tid + s]); __syncthreads(); }
  mx = red[0]; __syncthreads();
  float sum = 0.f;
  #pragma unroll
  for (int j = 0; j < 8; ++j) { v[j] = __expf(v[j] - mx); sum += v[j]; }
  red[tid] = sum; __syncthreads();
  for (int s = 128; s > 0; s >>= 1) { if (tid < s) red[tid] += red[tid + s]; __syncthreads(); }
  const float inv = 1.f / red[0];
  #pragma unroll
  for (int j = 0; j < 8; ++j) p[tid + j * 256] = v[j] * inv;
}

// ---------------------------------------------------------------- add + LayerNorm
__global__ __launch_bounds__(256)
void add_ln(const float* __restrict__ a, const float* __restrict__ b,
            const float* __restrict__ g, const float* __restrict__ beta,
            float* __restrict__ outf, bf16* __restrict__ out16) {
  __shared__ float red[256];
  const int row = blockIdx.x;
  const float* pa = a + (long long)row * D_MODEL;
  const float* pb = b + (long long)row * D_MODEL;
  const int tid = threadIdx.x;
  float v[4]; float s = 0.f;
  #pragma unroll
  for (int j = 0; j < 4; ++j) { const int i = tid + j * 256; v[j] = pa[i] + pb[i]; s += v[j]; }
  red[tid] = s; __syncthreads();
  for (int t = 128; t > 0; t >>= 1) { if (tid < t) red[tid] += red[tid + t]; __syncthreads(); }
  const float mu = red[0] * (1.f / D_MODEL); __syncthreads();
  float s2 = 0.f;
  #pragma unroll
  for (int j = 0; j < 4; ++j) { const float d = v[j] - mu; s2 += d * d; }
  red[tid] = s2; __syncthreads();
  for (int t = 128; t > 0; t >>= 1) { if (tid < t) red[tid] += red[tid + t]; __syncthreads(); }
  const float inv = rsqrtf(red[0] * (1.f / D_MODEL) + EPS);
  #pragma unroll
  for (int j = 0; j < 4; ++j) {
    const int i = tid + j * 256;
    const float y = (v[j] - mu) * inv * g[i] + beta[i];
    if (outf)  outf[(long long)row * D_MODEL + i] = y;
    if (out16) out16[(long long)row * D_MODEL + i] = (bf16)y;
  }
}

// ---------------------------------------------------------------- host
extern "C" void kernel_launch(void* const* d_in, const int* in_sizes, int n_in,
                              void* d_out, int out_size, void* d_ws, size_t ws_size,
                              hipStream_t stream) {
  (void)in_sizes; (void)n_in; (void)out_size; (void)ws_size;
  const float* x    = (const float*)d_in[0];
  const int*   mask = (const int*)d_in[1];
  const float* Wq   = (const float*)d_in[2];
  const float* bq   = (const float*)d_in[3];
  const float* Wk   = (const float*)d_in[4];
  const float* bk   = (const float*)d_in[5];
  const float* Wv   = (const float*)d_in[6];
  const float* bv   = (const float*)d_in[7];
  const float* Wo   = (const float*)d_in[8];
  const float* bo   = (const float*)d_in[9];
  const float* g1   = (const float*)d_in[10];
  const float* be1  = (const float*)d_in[11];
  const float* W1   = (const float*)d_in[12];
  const float* b1   = (const float*)d_in[13];
  const float* W2   = (const float*)d_in[14];
  const float* b2   = (const float*)d_in[15];
  const float* g2   = (const float*)d_in[16];
  const float* be2  = (const float*)d_in[17];

  float* out_f = (float*)d_out;
  float* attn  = out_f + (size_t)BATCH * SEQ * D_MODEL;

  char* ws = (char*)d_ws;
  const size_t MB = 1ull << 20;
  bf16* xb    = (bf16*)(ws + 0);        // dead after QKV -> recycled as ff1b
  bf16* qb    = (bf16*)(ws + 8  * MB);
  bf16* kb    = (bf16*)(ws + 16 * MB);
  bf16* vb    = (bf16*)(ws + 24 * MB);
  bf16* ff1b  = (bf16*)(ws + 0);        // overlays [0,32MB) after attention
  bf16* wqbT  = (bf16*)(ws + 32 * MB);
  bf16* wkbT  = (bf16*)(ws + 34 * MB);
  bf16* wvbT  = (bf16*)(ws + 36 * MB);
  bf16* wobT  = (bf16*)(ws + 38 * MB);
  bf16* w1bT  = (bf16*)(ws + 40 * MB);
  bf16* w2bT  = (bf16*)(ws + 48 * MB);
  bf16* ctxb  = (bf16*)(ws + 56 * MB);
  float* attn_out = (float*)(ws + 64 * MB);   // consumed by ln1,
  float* ff2f     = (float*)(ws + 64 * MB);   // region reused for FF2 out
  float* hf   = (float*)(ws + 80 * MB);
  bf16*  hb   = (bf16*)(ws + 96 * MB);
  bf16*  vT   = (bf16*)(ws + 104 * MB);       // [B,H,64,S]

  const int NT = BATCH * SEQ;                 // 4096 token rows
  dim3 blk(256);

  // x -> bf16; weights -> transposed bf16 [N][K]
  { int n = NT * D_MODEL; cvt_f32_bf16<<<n / 1024, blk, 0, stream>>>(x, xb, n); }
  {
    dim3 g(D_MODEL / 32, D_MODEL / 32);
    transpose_cvt_w<<<g, blk, 0, stream>>>(Wq, wqbT, D_MODEL, D_MODEL);
    transpose_cvt_w<<<g, blk, 0, stream>>>(Wk, wkbT, D_MODEL, D_MODEL);
    transpose_cvt_w<<<g, blk, 0, stream>>>(Wv, wvbT, D_MODEL, D_MODEL);
    transpose_cvt_w<<<g, blk, 0, stream>>>(Wo, wobT, D_MODEL, D_MODEL);
  }
  transpose_cvt_w<<<dim3(D_FF / 32, D_MODEL / 32), blk, 0, stream>>>(W1, w1bT, D_MODEL, D_FF);
  transpose_cvt_w<<<dim3(D_MODEL / 32, D_FF / 32), blk, 0, stream>>>(W2, w2bT, D_FF, D_MODEL);

  // Q, K, V projections: [4096,1024] @ WT^T + bias -> bf16
  {
    dim3 g(D_MODEL / 128, NT / 128, 1);
    gemm_wmma<128, false, false><<<g, blk, 0, stream>>>(
        xb, 0, 0, D_MODEL, wqbT, 0, 0, D_MODEL, bq, 1.f,
        nullptr, 0, 0, 0, qb, 0, 0, D_MODEL, D_MODEL);
    gemm_wmma<128, false, false><<<g, blk, 0, stream>>>(
        xb, 0, 0, D_MODEL, wkbT, 0, 0, D_MODEL, bk, 1.f,
        nullptr, 0, 0, 0, kb, 0, 0, D_MODEL, D_MODEL);
    gemm_wmma<128, false, false><<<g, blk, 0, stream>>>(
        xb, 0, 0, D_MODEL, wvbT, 0, 0, D_MODEL, bv, 1.f,
        nullptr, 0, 0, 0, vb, 0, 0, D_MODEL, D_MODEL);
  }
  // v -> vT [B,H,64,S]
  transpose_v<<<dim3(SEQ / 32, 2, BATCH * NUM_HEADS), blk, 0, stream>>>(vb, vT);
  // scores = (q @ k^T) / 8 per (b,h), f32 straight into d_out attn region
  {
    dim3 g(SEQ / 128, SEQ / 128, BATCH * NUM_HEADS);
    gemm_wmma<128, false, false><<<g, blk, 0, stream>>>(
        qb, (long long)SEQ * D_MODEL, 64, D_MODEL,
        kb, (long long)SEQ * D_MODEL, 64, D_MODEL,
        nullptr, 0.125f,
        attn, (long long)NUM_HEADS * SEQ * SEQ, (long long)SEQ * SEQ, SEQ,
        nullptr, 0, 0, 0, DK);
  }
  softmax_rows<<<BATCH * NUM_HEADS * SEQ, blk, 0, stream>>>(attn, mask);
  // ctx = attn @ v  (A f32 from d_out, B = vT [64][S])
  {
    dim3 g(1, SEQ / 128, BATCH * NUM_HEADS);
    gemm_wmma<64, true, false><<<g, blk, 0, stream>>>(
        attn, (long long)NUM_HEADS * SEQ * SEQ, (long long)SEQ * SEQ, SEQ,
        vT, (long long)NUM_HEADS * DK * SEQ, (long long)DK * SEQ, SEQ,
        nullptr, 1.f,
        nullptr, 0, 0, 0,
        ctxb, (long long)SEQ * D_MODEL, 64, D_MODEL,
        SEQ);
  }
  // attn_out = ctx @ Wo + bo (f32)
  {
    dim3 g(D_MODEL / 128, NT / 128, 1);
    gemm_wmma<128, false, false><<<g, blk, 0, stream>>>(
        ctxb, 0, 0, D_MODEL, wobT, 0, 0, D_MODEL, bo, 1.f,
        attn_out, 0, 0, D_MODEL, nullptr, 0, 0, 0, D_MODEL);
  }
  add_ln<<<NT, blk, 0, stream>>>(x, attn_out, g1, be1, hf, hb);
  // ff1 = relu(h @ W1 + b1) -> bf16
  {
    dim3 g(D_FF / 128, NT / 128, 1);
    gemm_wmma<128, false, true><<<g, blk, 0, stream>>>(
        hb, 0, 0, D_MODEL, w1bT, 0, 0, D_MODEL, b1, 1.f,
        nullptr, 0, 0, 0, ff1b, 0, 0, D_FF, D_MODEL);
  }
  // ff2 = ff1 @ W2 + b2 -> f32
  {
    dim3 g(D_MODEL / 128, NT / 128, 1);
    gemm_wmma<128, false, false><<<g, blk, 0, stream>>>(
        ff1b, 0, 0, D_FF, w2bT, 0, 0, D_FF, b2, 1.f,
        ff2f, 0, 0, D_MODEL, nullptr, 0, 0, 0, D_FF);
  }
  add_ln<<<NT, blk, 0, stream>>>(hf, ff2f, g2, be2, out_f, nullptr);
}
